// ProtICU_76476187672867
// MI455X (gfx1250) — matrix operand order repaired and checked by reading, not compiled, not run on hardware
//
#include <hip/hip_runtime.h>
#include <math.h>
#include <stdint.h>

// ---------------- types ----------------
typedef __attribute__((ext_vector_type(16))) __bf16          v16bf;
typedef __attribute__((ext_vector_type(16))) unsigned short  v16u;
typedef __attribute__((ext_vector_type(8)))  float           v8f;
typedef __attribute__((ext_vector_type(4)))  unsigned        v4u;
typedef __attribute__((ext_vector_type(8)))  unsigned        v8u;

__device__ __forceinline__ unsigned short f2bf(float f) {
    unsigned u = __float_as_uint(f);
    u += 0x7FFFu + ((u >> 16) & 1u);          // round-to-nearest-even
    return (unsigned short)(u >> 16);
}
__device__ __forceinline__ float bf2f(unsigned short h) {
    return __uint_as_float(((unsigned)h) << 16);
}

// ---------------- problem constants ----------------
#define BATCH 32
#define FEAT  76
#define LSEQ  2048
#define CH    128
#define PROTO 128
#define DDIM  128

// ---------------- prep: x (B,L,F) f32 -> h0 (B,F,L+4) bf16 with zero halo ----------------
__global__ void prep_x_kernel(const float* __restrict__ x, unsigned short* __restrict__ h0) {
    int idx = blockIdx.x * 256 + threadIdx.x;
    const int perB = FEAT * (LSEQ + 4);
    if (idx >= BATCH * perB) return;
    int b = idx / perB;
    int r = idx - b * perB;
    int c = r / (LSEQ + 4);
    int lp = r - c * (LSEQ + 4);
    unsigned short v = 0;
    if (lp >= 2 && lp < LSEQ + 2)
        v = f2bf(x[(b * LSEQ + (lp - 2)) * FEAT + c]);
    h0[idx] = v;
}

// ---------------- pack weights (O, C, KW) f32 -> A-fragment layout bf16 ----------------
// out[((w*ksteps + s)*32 + lane)*16 + i]; kpos = 16*(lane/16) + i (contiguous k order, matches B fill).
__global__ void pack_w_kernel(const float* __restrict__ W, unsigned short* __restrict__ out,
                              int C, int KW, int ksteps, int total) {
    int idx = blockIdx.x * 256 + threadIdx.x;
    if (idx >= total) return;
    int i    = idx & 15;
    int lane = (idx >> 4) & 31;
    int rest = idx >> 9;                  // = w*ksteps + s
    int s = rest % ksteps;
    int w = rest / ksteps;
    int m = w * 16 + (lane & 15);
    int kpos = (lane & 16) + i;
    int ck = s * 32 + kpos;
    float val = 0.f;
    int CK = C * KW;
    if (ck < CK) {
        int c  = ck / KW;
        int kk = ck - c * KW;
        val = W[(m * C + c) * KW + kk];
    }
    out[idx] = f2bf(val);
}

// ---------------- prototype norms (bf16-rounded, consistent with GEMM inputs) ----------------
__global__ void pnorm_kernel(const float* __restrict__ proto, float* __restrict__ pnorm) {
    int p = threadIdx.x;
    if (p >= PROTO) return;
    float s = 0.f;
    for (int d = 0; d < DDIM; ++d) {
        float v = bf2f(f2bf(proto[p * DDIM + d]));
        s += v * v;
    }
    pnorm[p] = s;
}

// ---------------- feature norms: g2 (B,128,256) bf16 -> fnorm (B,256) ----------------
__global__ void fnorm_kernel(const unsigned short* __restrict__ g2, float* __restrict__ fnorm) {
    int idx = blockIdx.x * 256 + threadIdx.x;
    if (idx >= BATCH * 256) return;
    int b = idx >> 8, l = idx & 255;
    float s = 0.f;
    for (int c = 0; c < CH; ++c) {
        float v = bf2f(g2[(b * CH + c) * 256 + l]);
        s += v * v;
    }
    fnorm[idx] = s;
}

__global__ void init_min_kernel(unsigned int* __restrict__ minbuf) {
    int idx = blockIdx.x * 256 + threadIdx.x;
    if (idx < BATCH * PROTO) minbuf[idx] = 0x7F800000u;   // +inf
}

// ---------------- unified WMMA GEMM kernel ----------------
// MODE 0: conv epilogue  -> bias+ReLU+maxpool2, store bf16 (B,128,LoutPad) at outHalo
// MODE 1: 1x1 epilogue   -> bias+ReLU, store bf16
// MODE 2: distance       -> d2 = fnorm - 2*acc + pnorm, sqrt, lane-min, atomicMin
// Block = 256 threads (8 waves). Wave w computes M-tile w (16 rows), shared 16-pos N-tile.
// Phase 1 staging: KW==5 -> TDM 2D tile DMA (tensor_load_to_lds, TENSORcnt);
//                  KW==1 -> per-lane async global->LDS B32 copies (ASYNCcnt).
template <int KW, int MODE>
__global__ void gemm_kernel(const unsigned short* __restrict__ in,
                            const unsigned short* __restrict__ wpack,
                            const float* __restrict__ bias,
                            unsigned short* __restrict__ out,
                            const float* __restrict__ fnorm,
                            const float* __restrict__ pnorm,
                            unsigned int* __restrict__ minbuf,
                            int C, int ksteps, int LinPad, int inOff,
                            int nTiles, int LoutPad, int outHalo) {
    constexpr int RAW = 16 + KW - 1;                 // raw tile width per channel
    const int CK     = C * KW;
    const int CKpad  = ksteps * 32;
    const int LDSTR  = CKpad + 16;                   // multiple of 16 -> 32B-aligned fragments
    extern __shared__ unsigned short smem[];
    unsigned short* raw    = smem;                   // C * RAW
    unsigned short* im2col = smem + C * RAW;         // 16 * LDSTR  (C*RAW*2 bytes is 32B-mult for our configs)

    const int tid  = threadIdx.x;
    const int b    = blockIdx.x / nTiles;
    const int t    = blockIdx.x - b * nTiles;
    const int l0   = t * 16;

    // ---- phase 1: raw tile staging (global -> LDS) via CDNA5 DMA paths ----
    if constexpr (KW == 5) {
        // One TDM descriptor: 2D tile, C rows x RAW bf16 elements, row stride LinPad.
        if (tid < 32) {                              // wave 0 issues the DMA
            unsigned long long ga =
                (unsigned long long)(uintptr_t)(in + (size_t)b * C * LinPad + inOff + l0);
            unsigned ldsb = (unsigned)(uintptr_t)raw;
            v4u g0;
            g0[0] = 1u;                               // count=1 (valid user descriptor)
            g0[1] = ldsb;                             // lds_addr (bytes)
            g0[2] = (unsigned)ga;                     // global_addr[31:0]
            g0[3] = ((unsigned)(ga >> 32) & 0x01FFFFFFu) | 0x80000000u; // addr[56:32] | type=2
            v8u g1;
            g1[0] = 0x00010000u;                      // data_size=1 (2 bytes), no mask/pad/iter
            g1[1] = ((unsigned)RAW) << 16;            // tensor_dim0[15:0] @ bits[63:48]
            g1[2] = ((unsigned)C) << 16;              // tensor_dim1[15:0] @ bits[95:80]
            g1[3] = ((unsigned)RAW) << 16;            // tile_dim0 @ bits[127:112]
            g1[4] = (unsigned)C;                      // tile_dim1 @ bits[143:128], tile_dim2=0
            g1[5] = (unsigned)LinPad;                 // tensor_dim0_stride[31:0]
            g1[6] = 0u;
            g1[7] = 0u;
            asm volatile("tensor_load_to_lds %0, %1" :: "s"(g0), "s"(g1) : "memory");
            __builtin_amdgcn_s_wait_tensorcnt(0);
        }
    } else {
        // KW==1: RAW=16 -> 8 dwords per channel row, rows are dword-aligned runs.
        const int dw = C * (RAW / 2);
        for (int e = tid; e < dw; e += 256) {
            int c  = e >> 3;                          // e / (RAW/2), RAW/2 == 8
            int jd = e & 7;
            const unsigned short* gp =
                in + ((size_t)b * C + c) * LinPad + inOff + l0 + jd * 2;
            unsigned la = (unsigned)(uintptr_t)(raw + e * 2);
            asm volatile("global_load_async_to_lds_b32 %0, %1, off"
                         :: "v"(la), "v"(gp) : "memory");
        }
        asm volatile("s_wait_asynccnt 0x0" ::: "memory");
    }
    __syncthreads();

    // ---- phase 2: im2col expansion (LDS -> LDS), zero-pad ck in [CK, CKpad) ----
    const int total2 = 16 * CKpad;
    for (int e = tid; e < total2; e += 256) {
        int ck = e >> 4;
        int n  = e & 15;
        unsigned short v = 0;
        if (ck < CK) {
            int c  = ck / KW;
            int kk = ck - c * KW;
            v = raw[c * RAW + n + kk];
        }
        im2col[n * LDSTR + ck] = v;
    }
    __syncthreads();

    // ---- WMMA main loop ----
    const int wave = tid >> 5;
    const int lane = tid & 31;
    v8f acc = {0.f, 0.f, 0.f, 0.f, 0.f, 0.f, 0.f, 0.f};
    const int nIdx = lane & 15;
    for (int s = 0; s < ksteps; ++s) {
        v16u av = *(const v16u*)(wpack + (((wave * ksteps + s) * 32 + lane) << 4));
        v16u bv = *(const v16u*)(im2col + nIdx * LDSTR + s * 32 + (lane & 16));
        acc = __builtin_amdgcn_wmma_f32_16x16x32_bf16(
                  false, __builtin_bit_cast(v16bf, av),
                  false, __builtin_bit_cast(v16bf, bv),
                  (short)0, acc, false, false);
    }

    const int hi = (lane >> 4) & 1;                  // lanes 16-31 hold rows r+8
    if constexpr (MODE == 2) {
        const int l = l0 + nIdx;
        const float fn = fnorm[b * LinPad + l];
        #pragma unroll
        for (int r = 0; r < 8; ++r) {
            int m = wave * 16 + r + (hi << 3);
            float d2  = fn - 2.f * acc[r] + pnorm[m];
            float dis = sqrtf(fmaxf(d2, 1e-12f));
            // min over the 16 positions (lanes within each 16-lane half)
            #pragma unroll
            for (int off = 1; off < 16; off <<= 1)
                dis = fminf(dis, __shfl_xor(dis, off, 32));
            if (nIdx == 0)
                atomicMin(&minbuf[b * PROTO + m], __float_as_uint(dis));
        }
    } else {
        #pragma unroll
        for (int r = 0; r < 8; ++r) {
            int m = wave * 16 + r + (hi << 3);
            float v = fmaxf(acc[r] + bias[m], 0.f);
            if constexpr (MODE == 0) {
                float o = fmaxf(v, __shfl_xor(v, 1, 32));
                if ((lane & 1) == 0)
                    out[(b * CH + m) * LoutPad + outHalo + ((l0 + nIdx) >> 1)] = f2bf(o);
            } else {
                out[(b * CH + m) * LoutPad + outHalo + l0 + nIdx] = f2bf(v);
            }
        }
    }
}

// ---------------- final head: min_dis -> sim -> sigmoid(sim @ lw^T) ----------------
// one wave per batch row
__global__ void final_kernel(const unsigned int* __restrict__ minbuf,
                             const int* __restrict__ classes,
                             float* __restrict__ d_out) {
    int b = blockIdx.x;
    int lane = threadIdx.x;
    float s0 = 0.f, s1 = 0.f;
    for (int j = 0; j < 4; ++j) {
        int p = lane + 32 * j;
        float md = __uint_as_float(minbuf[b * PROTO + p]);
        d_out[2 * 32 + b * PROTO + p] = md;                 // min_dis output after `out`
        float sim = logf((md + 1.0f) / (md + 1e-4f));
        int cls = classes[p];
        s0 += sim * (cls == 0 ? 1.0f : -0.5f);
        s1 += sim * (cls == 1 ? 1.0f : -0.5f);
    }
    #pragma unroll
    for (int off = 16; off >= 1; off >>= 1) {
        s0 += __shfl_xor(s0, off, 32);
        s1 += __shfl_xor(s1, off, 32);
    }
    if (lane == 0) {
        d_out[b * 2 + 0] = 1.0f / (1.0f + expf(-s0));
        d_out[b * 2 + 1] = 1.0f / (1.0f + expf(-s1));
    }
}

// ---------------- launcher ----------------
extern "C" void kernel_launch(void* const* d_in, const int* in_sizes, int n_in,
                              void* d_out, int out_size, void* d_ws, size_t ws_size,
                              hipStream_t stream) {
    const float* x    = (const float*)d_in[0];
    const float* W1   = (const float*)d_in[1];
    const float* b1   = (const float*)d_in[2];
    const float* W2   = (const float*)d_in[3];
    const float* b2   = (const float*)d_in[4];
    const float* W3   = (const float*)d_in[5];
    const float* b3   = (const float*)d_in[6];
    const float* Wo1  = (const float*)d_in[7];
    const float* bo1  = (const float*)d_in[8];
    const float* Wo2  = (const float*)d_in[9];
    const float* bo2  = (const float*)d_in[10];
    const float* prot = (const float*)d_in[11];
    const int*   cls  = (const int*)d_in[12];
    float* out = (float*)d_out;

    // workspace carve-up (all offsets 256B-aligned)
    char* base = (char*)d_ws;
    size_t off = 0;
    auto alloc = [&](size_t bytes) { void* p = base + off; off += (bytes + 255) & ~size_t(255); return p; };
    unsigned short* h0  = (unsigned short*)alloc((size_t)BATCH * FEAT * (LSEQ + 4) * 2);
    unsigned short* h1  = (unsigned short*)alloc((size_t)BATCH * CH * 1028 * 2);
    unsigned short* h2  = (unsigned short*)alloc((size_t)BATCH * CH * 516 * 2);
    unsigned short* h3  = (unsigned short*)alloc((size_t)BATCH * CH * 260 * 2);
    unsigned short* g1  = (unsigned short*)alloc((size_t)BATCH * CH * 256 * 2);
    unsigned short* g2  = (unsigned short*)alloc((size_t)BATCH * CH * 256 * 2);
    unsigned short* wp1 = (unsigned short*)alloc((size_t)8 * 12 * 512 * 2);
    unsigned short* wp2 = (unsigned short*)alloc((size_t)8 * 20 * 512 * 2);
    unsigned short* wp3 = (unsigned short*)alloc((size_t)8 * 20 * 512 * 2);
    unsigned short* wo1 = (unsigned short*)alloc((size_t)8 * 4 * 512 * 2);
    unsigned short* wo2 = (unsigned short*)alloc((size_t)8 * 4 * 512 * 2);
    unsigned short* pp  = (unsigned short*)alloc((size_t)8 * 4 * 512 * 2);
    float*          pn  = (float*)alloc((size_t)PROTO * 4);
    float*          fn  = (float*)alloc((size_t)BATCH * 256 * 4);
    unsigned int*   mn  = (unsigned int*)alloc((size_t)BATCH * PROTO * 4);

    auto cdiv = [](int a, int b) { return (a + b - 1) / b; };

    // --- prep ---
    prep_x_kernel<<<cdiv(BATCH * FEAT * (LSEQ + 4), 256), 256, 0, stream>>>(x, h0);
    { int tot = 8 * 12 * 512; pack_w_kernel<<<cdiv(tot,256),256,0,stream>>>(W1,  wp1, FEAT, 5, 12, tot); }
    { int tot = 8 * 20 * 512; pack_w_kernel<<<cdiv(tot,256),256,0,stream>>>(W2,  wp2, CH,   5, 20, tot); }
    { int tot = 8 * 20 * 512; pack_w_kernel<<<cdiv(tot,256),256,0,stream>>>(W3,  wp3, CH,   5, 20, tot); }
    { int tot = 8 *  4 * 512; pack_w_kernel<<<cdiv(tot,256),256,0,stream>>>(Wo1, wo1, CH,   1,  4, tot); }
    { int tot = 8 *  4 * 512; pack_w_kernel<<<cdiv(tot,256),256,0,stream>>>(Wo2, wo2, CH,   1,  4, tot); }
    { int tot = 8 *  4 * 512; pack_w_kernel<<<cdiv(tot,256),256,0,stream>>>(prot, pp, DDIM, 1,  4, tot); }
    pnorm_kernel<<<1, 128, 0, stream>>>(prot, pn);
    init_min_kernel<<<cdiv(BATCH * PROTO, 256), 256, 0, stream>>>(mn);

    // --- conv stack (KW=5, pool epilogue, TDM staging) ---
    // conv1: C=76, ksteps=12, Lin=2048 (pad 2052) -> h1 (pad 1028, halo 2)
    {
        size_t sh = ((size_t)FEAT * 20 + 16 * (12 * 32 + 16)) * 2;
        gemm_kernel<5, 0><<<BATCH * 128, 256, sh, stream>>>(
            h0, wp1, b1, h1, nullptr, nullptr, nullptr,
            FEAT, 12, LSEQ + 4, 0, 128, 1028, 2);
    }
    // conv2: C=128, ksteps=20, Lin=1024 (pad 1028) -> h2 (pad 516, halo 2)
    {
        size_t sh = ((size_t)CH * 20 + 16 * (20 * 32 + 16)) * 2;
        gemm_kernel<5, 0><<<BATCH * 64, 256, sh, stream>>>(
            h1, wp2, b2, h2, nullptr, nullptr, nullptr,
            CH, 20, 1028, 0, 64, 516, 2);
    }
    // conv3: Lin=512 (pad 516) -> h3 (pad 260, halo 2)
    {
        size_t sh = ((size_t)CH * 20 + 16 * (20 * 32 + 16)) * 2;
        gemm_kernel<5, 0><<<BATCH * 32, 256, sh, stream>>>(
            h2, wp3, b3, h3, nullptr, nullptr, nullptr,
            CH, 20, 516, 0, 32, 260, 2);
    }

    // --- 1x1 convs (KW=1, plain epilogue, async->LDS staging) ---
    {
        size_t sh = ((size_t)CH * 16 + 16 * (4 * 32 + 16)) * 2;
        gemm_kernel<1, 1><<<BATCH * 16, 256, sh, stream>>>(
            h3, wo1, bo1, g1, nullptr, nullptr, nullptr,
            CH, 4, 260, 2, 16, 256, 0);
        gemm_kernel<1, 1><<<BATCH * 16, 256, sh, stream>>>(
            g1, wo2, bo2, g2, nullptr, nullptr, nullptr,
            CH, 4, 256, 0, 16, 256, 0);
    }

    // --- distance ---
    fnorm_kernel<<<cdiv(BATCH * 256, 256), 256, 0, stream>>>(g2, fn);
    {
        size_t sh = ((size_t)CH * 16 + 16 * (4 * 32 + 16)) * 2;
        gemm_kernel<1, 2><<<BATCH * 16, 256, sh, stream>>>(
            g2, pp, nullptr, nullptr, fn, pn, mn,
            CH, 4, 256, 0, 16, 256, 0);
    }

    // --- head ---
    final_kernel<<<BATCH, 32, 0, stream>>>(mn, cls, out);
}
